// BiLSTM_23905787969894
// MI455X (gfx1250) — matrix operand order repaired and checked by reading, not compiled, not run on hardware
//
#include <hip/hip_runtime.h>
#include <hip/hip_bf16.h>
#include <stdint.h>

// ---------------------------------------------------------------------------
// BiLSTM-CRF for MI455X (gfx1250), wave32 + v_wmma_f32_16x16x32_bf16.
//   B=64 S=512 WV=300 K=128 C=9 D=H=600, 4H=2400
// Pipeline:
//   prep kernels  : f32->bf16 weight/input conversion with padding
//   k_kn_gemm     : WMMA  kn = knowledge @ ke_W^T + ke_b  -> x16[S][B][608]
//   k_lstm        : persistent per-(batch-slice,dir) scan; fused
//                   [h|x_t] @ [Whh|Wih]^T (K=1216) via WMMA, c in VGPRs,
//                   h in LDS; x_t slab staged via GLOBAL_LOAD_ASYNC_TO_LDS
//   k_dense       : WMMA  logits = h16 @ dense_W^T + b  -> d_out (f32)
//   k_viterbi     : scalar CRF decode -> int32 tags after logits in d_out
// ---------------------------------------------------------------------------

#define Bn 64
#define Sn 512
#define WVn 300
#define Kn 128
#define Cn 9
#define Hn 600
#define KP 608          // padded hidden/input width
#define KCAT 1216       // K of fused recurrent GEMM = 608(h) + 608(x)
#define G4P 2432        // 4 gates x 608 padded output rows
#define NJT 38          // j-tiles of 16 over 608
#define ROWS (Bn*Sn)    // 32768

// workspace byte offsets (total 140,416,000 B ~= 134 MB)
#define OFF_KB16  0u            //  knowledge bf16 [32768][128]
#define OFF_KEW   8388608u      //  ke_W bf16 [304][128] (rows>=300 zero)
#define OFF_X16   8466432u      //  x bf16 time-major [512][64][608]
#define OFF_WCAT  48312320u     //  [2 dirs][2432][1216] = [Whh | Wih] bf16
#define OFF_BIAS  60141568u     //  [2 dirs][2432] f32 (bih+bhh, pad 0)
#define OFF_DW    60161024u     //  dense_W bf16 [16][1216] (pad 0)
#define OFF_H16   60199936u     //  h bf16 [32768][1216] (row=b*S+s)
#define OFF_BP    139891712u    //  viterbi backptrs u8 [512][64][16]

typedef __attribute__((ext_vector_type(16))) __bf16 bf16x16;
typedef __attribute__((ext_vector_type(8)))  float  f32x8;
typedef int v4i32 __attribute__((vector_size(16)));

union FragAB { uint4 q[2]; bf16x16 v; };
union FragC  { f32x8 v; float f[8]; };

#if defined(__AMDGCN__) && __has_builtin(__builtin_amdgcn_global_load_async_to_lds_b128)
#define HAVE_ASYNC_LDS 1
#else
#define HAVE_ASYNC_LDS 0
#endif

static __device__ __forceinline__ void wait_asynccnt0() {
#if defined(__AMDGCN__)
#if __has_builtin(__builtin_amdgcn_s_wait_asynccnt)
    __builtin_amdgcn_s_wait_asynccnt(0);
#else
    asm volatile("s_wait_asynccnt 0x0" ::: "memory");
#endif
#endif
}

static __device__ __forceinline__ unsigned short f2bf(float x) {
    union { float f; uint32_t u; } v; v.f = x;
    uint32_t r = v.u + 0x7FFFu + ((v.u >> 16) & 1u);   // round-to-nearest-even
    return (unsigned short)(r >> 16);
}
static __device__ __forceinline__ float sigmf(float x) {
    return 1.0f / (1.0f + __expf(-x));
}

#define WMMA_BF16(acc, a, b) \
    __builtin_amdgcn_wmma_f32_16x16x32_bf16(false, (a), false, (b), (short)0, (acc), false, false)

// ------------------------------- prep kernels ------------------------------

__global__ void k_cvt_kb16(const float* __restrict__ kn, unsigned short* __restrict__ o) {
    int i = blockIdx.x * 256 + threadIdx.x;            // 32768*128
    o[i] = f2bf(kn[i]);
}

__global__ void k_cvt_keW(const float* __restrict__ w, unsigned short* __restrict__ o) {
    int i = blockIdx.x * 256 + threadIdx.x;            // 304*128
    int r = i / Kn, c = i % Kn;
    o[i] = (r < WVn) ? f2bf(w[r * Kn + c]) : 0;
}

__global__ void k_cvt_wcat(const float* __restrict__ wihF, const float* __restrict__ whhF,
                           const float* __restrict__ wihB, const float* __restrict__ whhB,
                           unsigned short* __restrict__ o) {
    int i = blockIdx.x * 256 + threadIdx.x;            // 2*2432*1216
    int dir = i / (G4P * KCAT);
    int rem = i - dir * (G4P * KCAT);
    int row = rem / KCAT, col = rem - row * KCAT;
    int g = row / KP, j = row - g * KP;
    const float* wih = dir ? wihB : wihF;
    const float* whh = dir ? whhB : whhF;
    float v = 0.0f;
    if (j < Hn) {
        int nr = g * Hn + j;
        if (col < KP) { if (col < Hn) v = whh[(size_t)nr * Hn + col]; }
        else          { int k = col - KP; if (k < Hn) v = wih[(size_t)nr * Hn + k]; }
    }
    o[i] = f2bf(v);
}

__global__ void k_bias(const float* __restrict__ bihF, const float* __restrict__ bhhF,
                       const float* __restrict__ bihB, const float* __restrict__ bhhB,
                       float* __restrict__ o) {
    int i = blockIdx.x * 256 + threadIdx.x;            // 2*2432
    int dir = i / G4P, row = i % G4P;
    int g = row / KP, j = row - g * KP;
    float v = 0.0f;
    if (j < Hn) {
        int nr = g * Hn + j;
        v = dir ? (bihB[nr] + bhhB[nr]) : (bihF[nr] + bhhF[nr]);
    }
    o[i] = v;
}

__global__ void k_cvt_dW(const float* __restrict__ w, unsigned short* __restrict__ o) {
    int i = blockIdx.x * 256 + threadIdx.x;            // 16*1216
    int r = i / KCAT, c = i - r * KCAT;
    o[i] = (r < Cn && c < 2 * Hn) ? f2bf(w[r * (2 * Hn) + c]) : 0;
}

__global__ void k_xbase(const float* __restrict__ data, unsigned short* __restrict__ x16) {
    int i = blockIdx.x * 256 + threadIdx.x;            // 32768*308
    int row = i / 308, c = i - row * 308;              // x16 row = t*64+b
    int t = row / Bn, b = row - t * Bn;
    if (c < WVn) x16[(size_t)row * KP + c] = f2bf(data[((size_t)b * Sn + t) * WVn + c]);
    else         x16[(size_t)row * KP + Hn + (c - WVn)] = 0;   // pad cols 600..607
}

__global__ void k_h16pad(unsigned short* __restrict__ h16) {
    int i = blockIdx.x * 256 + threadIdx.x;            // 32768*16
    int r = i / 16, c = i - r * 16;
    h16[(size_t)r * KCAT + 2 * Hn + c] = 0;            // pad cols 1200..1215
}

// -------------------- knowledge embedding GEMM (WMMA) ----------------------
// kn[B*S,300] = knowledge_bf16[B*S,128] @ keW^T + ke_b, scattered into
// x16[t*64+b][300+n].  2048 M-tiles x 19 N-tiles, one tile per wave.

__global__ void k_kn_gemm(const unsigned short* __restrict__ a16,
                          const unsigned short* __restrict__ w16,
                          const float* __restrict__ keb,
                          unsigned short* __restrict__ x16) {
    int wave = blockIdx.x * 8 + (threadIdx.x >> 5);
    if (wave >= 2048 * 19) return;
    int mt = wave / 19, nt = wave - mt * 19;
    int lane = threadIdx.x & 31, half = lane >> 4, l16 = lane & 15;
    int n = nt * 16 + l16;

    FragC acc;
    float bv = (n < WVn) ? keb[n] : 0.0f;
#pragma unroll
    for (int r = 0; r < 8; ++r) acc.f[r] = bv;

    const unsigned short* ab = a16 + (size_t)(mt * 16 + l16) * Kn;
    const unsigned short* bb = w16 + (size_t)n * Kn;
#pragma unroll
    for (int ks = 0; ks < 4; ++ks) {
        int k0 = ks * 32;
        FragAB a, b;
        a.q[0] = *(const uint4*)(ab + k0 + half * 8);
        a.q[1] = *(const uint4*)(ab + k0 + 16 + half * 8);
        b.q[0] = *(const uint4*)(bb + k0 + half * 8);
        b.q[1] = *(const uint4*)(bb + k0 + 16 + half * 8);
        acc.v = WMMA_BF16(acc.v, a.v, b.v);
    }
    if (n < WVn) {
#pragma unroll
        for (int r = 0; r < 8; ++r) {
            int gr = mt * 16 + r + half * 8;           // = b*S+s
            int b_ = gr >> 9, s_ = gr & 511;
            x16[((size_t)s_ * Bn + b_) * KP + WVn + n] = f2bf(acc.f[r]);
        }
    }
}

// --------------------------- fused BiLSTM scan -----------------------------
// grid (4 batch-slices, 2 dirs), 1024 threads (32 waves).  Per step:
// G = bias + [h_prev | x_t] @ Wcat^T  via 16x16x32 bf16 WMMA, K=1216.
// Each wave owns j-tiles {w, w+32} (<38): 4 gate accumulators each,
// cell state c kept in VGPRs for all 512 steps, h in LDS (bf16).
// x_t slab (19.5 KB) is DMA'd to LDS with GLOBAL_LOAD_ASYNC_TO_LDS_B128
// (ASYNCcnt), drained and barriered before the WMMA weight stream starts.

__global__ __launch_bounds__(1024) void k_lstm(const unsigned short* __restrict__ x16,
                                               const unsigned short* __restrict__ wcat,
                                               const float* __restrict__ bias,
                                               unsigned short* __restrict__ h16) {
    __shared__ __align__(16) unsigned short A[16 * KCAT];  // [h(608) | x(608)] bf16
    const int bs  = blockIdx.x;       // batch slice: rows bs*16 .. bs*16+15
    const int dir = blockIdx.y;       // 0 forward, 1 backward
    const int tid = threadIdx.x;
    const int w   = tid >> 5;
    const int lane = tid & 31, half = lane >> 4, l16 = lane & 15;

    const unsigned short* wc = wcat + (size_t)dir * G4P * KCAT;
    const float* bi = bias + dir * G4P;

    for (int i = tid; i < 16 * KCAT; i += 1024) A[i] = 0;   // h0 = 0, pads = 0
    __syncthreads();

    float cst[2][8];
#pragma unroll
    for (int s = 0; s < 2; ++s)
#pragma unroll
        for (int r = 0; r < 8; ++r) cst[s][r] = 0.0f;

    for (int tt = 0; tt < Sn; ++tt) {
        const int t = dir ? (Sn - 1 - tt) : tt;

        // ---- stage x_t slab (16 rows x 608 bf16, contiguous in x16) -> LDS
#if HAVE_ASYNC_LDS
        // 16 rows x 76 x b128 chunks, per-lane async DMA (no VGPR bounce)
        for (int i = tid; i < 16 * 76; i += 1024) {
            int row = i / 76, c = i - row * 76;
            const unsigned short* gp =
                x16 + (size_t)(t * Bn + bs * 16 + row) * KP + c * 8;
            unsigned short* lp = (unsigned short*)&A[row * KCAT + KP + c * 8];
            __builtin_amdgcn_global_load_async_to_lds_b128(
                (__attribute__((address_space(1))) v4i32*)gp,
                (__attribute__((address_space(3))) v4i32*)lp,
                /*offset=*/0, /*cpol=*/0);
        }
#else
        const uint32_t* src = (const uint32_t*)x16 + ((size_t)(t * Bn + bs * 16) * KP >> 1);
        uint32_t* dst = (uint32_t*)A;
        for (int i = tid; i < 16 * 304; i += 1024) {
            int row = i / 304, cd = i - row * 304;
            dst[row * (KCAT / 2) + 304 + cd] = src[row * (KP / 2) + cd];
        }
#endif
        if (tt < Sn - 1) {                       // prefetch next step's slab
            const int tn = dir ? (t - 1) : (t + 1);
            const char* np = (const char*)(x16 + (size_t)(tn * Bn + bs * 16) * KP);
            if (tid < 76) __builtin_prefetch(np + tid * 256, 0, 1);
        }
#if HAVE_ASYNC_LDS
        wait_asynccnt0();                        // my async writes landed in LDS
#endif
        __syncthreads();                         // everyone's writes visible

        unsigned short hval[2][8];
#pragma unroll
        for (int si = 0; si < 2; ++si) {
            int jt = w + si * 32;
            if (jt >= NJT) continue;
            FragC acc[4];
#pragma unroll
            for (int g = 0; g < 4; ++g) {
                float bv = bi[g * KP + jt * 16 + l16];
#pragma unroll
                for (int r = 0; r < 8; ++r) acc[g].f[r] = bv;
            }
            for (int ks = 0; ks < KCAT / 32; ++ks) {   // 38 k-steps
                int k0 = ks * 32;
                FragAB a;
                const unsigned short* ap = A + (l16 * KCAT + k0 + half * 8);
                a.q[0] = *(const uint4*)ap;            // ds_load_b128
                a.q[1] = *(const uint4*)(ap + 16);
#pragma unroll
                for (int g = 0; g < 4; ++g) {          // reuse A across 4 gates
                    FragAB b;
                    const unsigned short* bp = wc +
                        ((size_t)(g * KP + jt * 16 + l16) * KCAT + k0 + half * 8);
                    b.q[0] = *(const uint4*)bp;
                    b.q[1] = *(const uint4*)(bp + 16);
                    acc[g].v = WMMA_BF16(acc[g].v, a.v, b.v);
                }
            }
            // ---- elementwise LSTM cell (pytorch gate order i,f,g,o)
#pragma unroll
            for (int r = 0; r < 8; ++r) {
                float i_ = sigmf(acc[0].f[r]);
                float f_ = sigmf(acc[1].f[r]);
                float g_ = tanhf(acc[2].f[r]);
                float o_ = sigmf(acc[3].f[r]);
                float c  = f_ * cst[si][r] + i_ * g_;
                cst[si][r] = c;
                hval[si][r] = f2bf(o_ * tanhf(c));
            }
        }
        __syncthreads();   // all WMMA reads of A done before overwriting h

#pragma unroll
        for (int si = 0; si < 2; ++si) {
            int jt = w + si * 32;
            if (jt >= NJT) continue;
            int j = jt * 16 + l16;
            if (j >= Hn) continue;
#pragma unroll
            for (int r = 0; r < 8; ++r) {
                int m = r + half * 8;
                unsigned short hv = hval[si][r];
                A[m * KCAT + j] = hv;                                     // next step
                h16[((size_t)(bs * 16 + m) * Sn + t) * KCAT + dir * Hn + j] = hv;
            }
        }
        // loop-top staging touches only x columns (>=608): no conflict with
        // these h writes; the post-stage barrier makes h visible to all waves.
    }

    // Where an N-split (cluster) variant would exchange h slices each step;
    // architecturally a NOP when not dispatched as a workgroup cluster
    // (ClusterID==0), so semantics are unchanged here.
#if __has_builtin(__builtin_amdgcn_s_cluster_barrier)
    __builtin_amdgcn_s_cluster_barrier();
#endif
}

// ------------------------------ dense head ---------------------------------
// logits[B*S,9] = h16[B*S,1216] @ dW16[16,1216]^T + dense_b   (one N-tile)

__global__ void k_dense(const unsigned short* __restrict__ h16,
                        const unsigned short* __restrict__ dw,
                        const float* __restrict__ db,
                        float* __restrict__ logits) {
    int mt = blockIdx.x * 8 + (threadIdx.x >> 5);
    if (mt >= ROWS / 16) return;
    int lane = threadIdx.x & 31, half = lane >> 4, l16 = lane & 15;
    int n = l16;

    FragC acc;
    float bv = (n < Cn) ? db[n] : 0.0f;
#pragma unroll
    for (int r = 0; r < 8; ++r) acc.f[r] = bv;

    const unsigned short* ab = h16 + (size_t)(mt * 16 + l16) * KCAT;
    const unsigned short* bb = dw + (size_t)n * KCAT;
    for (int ks = 0; ks < KCAT / 32; ++ks) {
        int k0 = ks * 32;
        FragAB a, b;
        a.q[0] = *(const uint4*)(ab + k0 + half * 8);
        a.q[1] = *(const uint4*)(ab + k0 + 16 + half * 8);
        b.q[0] = *(const uint4*)(bb + k0 + half * 8);
        b.q[1] = *(const uint4*)(bb + k0 + 16 + half * 8);
        acc.v = WMMA_BF16(acc.v, a.v, b.v);
    }
    if (n < Cn) {
#pragma unroll
        for (int r = 0; r < 8; ++r) {
            int gr = mt * 16 + r + half * 8;           // row = b*S+s
            logits[(size_t)gr * Cn + n] = acc.f[r];
        }
    }
}

// ----------------------------- CRF Viterbi ---------------------------------
// one thread per batch element; 9 states; bp in workspace (u8).

__global__ void k_viterbi(const float* __restrict__ logits,
                          const unsigned char* __restrict__ mask,
                          const float* __restrict__ trans,
                          const float* __restrict__ start_t,
                          const float* __restrict__ end_t,
                          unsigned char* __restrict__ bp,
                          int* __restrict__ dec) {
    __shared__ float sT[Cn * Cn], sS[Cn], sE[Cn];
    int tid = threadIdx.x;
    if (tid < Cn * Cn) sT[tid] = trans[tid];
    if (tid < Cn) { sS[tid] = start_t[tid]; sE[tid] = end_t[tid]; }
    __syncthreads();
    if (tid >= Bn) return;
    const int b = tid;

    float score[Cn];
#pragma unroll
    for (int c = 0; c < Cn; ++c) score[c] = sS[c] + logits[((size_t)b * Sn) * Cn + c];

    for (int s = 1; s < Sn; ++s) {
        float em[Cn];
#pragma unroll
        for (int c = 0; c < Cn; ++c) em[c] = logits[((size_t)b * Sn + s) * Cn + c];
        unsigned char m = mask[b * Sn + s];
        float ns[Cn]; int arg[Cn];
#pragma unroll
        for (int cn = 0; cn < Cn; ++cn) {
            float best = -3.0e38f; int a = 0;
#pragma unroll
            for (int cp = 0; cp < Cn; ++cp) {
                float v = score[cp] + sT[cp * Cn + cn];
                if (v > best) { best = v; a = cp; }    // first-max, like argmax
            }
            ns[cn] = best + em[cn];
            arg[cn] = m ? a : cn;                      // identity through pad
        }
        if (m) {
#pragma unroll
            for (int cn = 0; cn < Cn; ++cn) score[cn] = ns[cn];
        }
#pragma unroll
        for (int cn = 0; cn < Cn; ++cn) bp[((size_t)s * Bn + b) * 16 + cn] = (unsigned char)arg[cn];
    }

    float best = -3.0e38f; int tag = 0;
#pragma unroll
    for (int c = 0; c < Cn; ++c) {
        float v = score[c] + sE[c];
        if (v > best) { best = v; tag = c; }
    }
    dec[b * Sn + (Sn - 1)] = mask[b * Sn + (Sn - 1)] ? tag : 0;
    for (int s = Sn - 2; s >= 0; --s) {
        tag = bp[((size_t)(s + 1) * Bn + b) * 16 + tag];
        dec[b * Sn + s] = mask[b * Sn + s] ? tag : 0;
    }
}

// ------------------------------- launcher ----------------------------------

extern "C" void kernel_launch(void* const* d_in, const int* in_sizes, int n_in,
                              void* d_out, int out_size, void* d_ws, size_t ws_size,
                              hipStream_t stream) {
    (void)in_sizes; (void)n_in; (void)out_size; (void)ws_size;
    const float* data      = (const float*)d_in[0];
    const float* knowledge = (const float*)d_in[1];
    const unsigned char* mask = (const unsigned char*)d_in[2];
    const float* keW   = (const float*)d_in[3];
    const float* keb   = (const float*)d_in[4];
    const float* wihF  = (const float*)d_in[5];
    const float* whhF  = (const float*)d_in[6];
    const float* bihF  = (const float*)d_in[7];
    const float* bhhF  = (const float*)d_in[8];
    const float* wihB  = (const float*)d_in[9];
    const float* whhB  = (const float*)d_in[10];
    const float* bihB  = (const float*)d_in[11];
    const float* bhhB  = (const float*)d_in[12];
    const float* dW    = (const float*)d_in[13];
    const float* db    = (const float*)d_in[14];
    const float* trans = (const float*)d_in[15];
    const float* st_t  = (const float*)d_in[16];
    const float* en_t  = (const float*)d_in[17];

    char* ws = (char*)d_ws;
    unsigned short* kb16  = (unsigned short*)(ws + OFF_KB16);
    unsigned short* keW16 = (unsigned short*)(ws + OFF_KEW);
    unsigned short* x16   = (unsigned short*)(ws + OFF_X16);
    unsigned short* wcat  = (unsigned short*)(ws + OFF_WCAT);
    float*          biasb = (float*)(ws + OFF_BIAS);
    unsigned short* dW16  = (unsigned short*)(ws + OFF_DW);
    unsigned short* h16   = (unsigned short*)(ws + OFF_H16);
    unsigned char*  bpbuf = (unsigned char*)(ws + OFF_BP);

    float* logits = (float*)d_out;
    int*   dec    = (int*)d_out + (size_t)Bn * Sn * Cn;

    // prep
    k_cvt_kb16<<<16384, 256, 0, stream>>>(knowledge, kb16);
    k_cvt_keW <<<152,   256, 0, stream>>>(keW, keW16);
    k_cvt_wcat<<<23104, 256, 0, stream>>>(wihF, whhF, wihB, whhB, wcat);
    k_bias    <<<19,    256, 0, stream>>>(bihF, bhhF, bihB, bhhB, biasb);
    k_cvt_dW  <<<76,    256, 0, stream>>>(dW, dW16);
    k_xbase   <<<39424, 256, 0, stream>>>(data, x16);
    k_h16pad  <<<2048,  256, 0, stream>>>(h16);
    // knowledge embedding GEMM (fills x16 cols 300..599)
    k_kn_gemm <<<4864,  256, 0, stream>>>(kb16, keW16, keb, x16);
    // fused bidirectional LSTM scan
    k_lstm    <<<dim3(4, 2), 1024, 0, stream>>>(x16, wcat, biasb, h16);
    // dense head -> logits (d_out)
    k_dense   <<<256,   256, 0, stream>>>(h16, dW16, db, logits);
    // CRF decode -> int32 tags after logits
    k_viterbi <<<1,     128, 0, stream>>>(logits, mask, trans, st_t, en_t, bpbuf, dec);
}
